// LinearAttention_41575283425767
// MI455X (gfx1250) — compile-verified, matrix operand order checked by reading, not run on hardware
//
#include <hip/hip_runtime.h>

// ---------------- types ----------------
typedef __attribute__((ext_vector_type(16))) __bf16 v16bf;
typedef __attribute__((ext_vector_type(8)))  float  v8f;
typedef unsigned int v4u __attribute__((ext_vector_type(4)));
typedef int          v8i __attribute__((ext_vector_type(8)));
typedef int          v4i __attribute__((ext_vector_type(4)));

#define LSEQ   2048
#define NHEAD  16
#define FD     16
#define HDIM   64
#define DPAD   288      // 273 features padded to 9*32
#define CHUNK  32
#define NCHUNK (LSEQ / CHUNK)

// ---------------- helpers ----------------
__device__ __forceinline__ unsigned short f2bf(float f) {
    union { float f; unsigned u; } v; v.f = f;
    unsigned r = v.u + 0x7FFFu + ((v.u >> 16) & 1u);
    return (unsigned short)(r >> 16);
}
__device__ __forceinline__ float bf2f(unsigned short u) {
    union { unsigned u; float f; } v; v.u = ((unsigned)u) << 16;
    return v.f;
}

// generic LDS pointer -> byte offset within workgroup LDS segment
typedef __attribute__((address_space(3))) unsigned char lds_byte_t;
__device__ __forceinline__ unsigned lds_offset(const void* p) {
    return (unsigned)(unsigned long long)(lds_byte_t*)(p);
}

// Load a 16x32 bf16 A-fragment (row-major source, stride in elements).
__device__ __forceinline__ v16bf load_frag(const unsigned short* base, int stride, int lane) {
    int row = lane & 15;
    int kh  = (lane >> 4) * 8;
    const unsigned short* p = base + row * stride + kh;
    union { v16bf v; unsigned u[8]; } r;
#pragma unroll
    for (int j = 0; j < 4; ++j) {
        r.u[j]     = *(const unsigned*)(p + 2 * j);
        r.u[4 + j] = *(const unsigned*)(p + 16 + 2 * j);
    }
    return r.v;
}

__device__ __forceinline__ v8f wmma_bf16(v16bf a, v16bf b, v8f c) {
    return __builtin_amdgcn_wmma_f32_16x16x32_bf16(false, a, false, b, (short)0, c, false, false);
}

// ---------------- Tensor Data Mover: 2D tile  global -> LDS ----------------
// data_size_log: 0=1B,1=2B,2=4B,3=8B ; dims/strides in data_size units
__device__ __forceinline__ void tdm_load_2d(unsigned lds_off, unsigned long long gaddr,
                                            unsigned tensor_d0, unsigned tensor_d1,
                                            unsigned tile_d0, unsigned tile_d1,
                                            unsigned long long stride0, int data_size_log) {
    v4u g0;
    g0.x = 1u;                                                   // count=1, user mode
    g0.y = lds_off;                                              // lds_addr
    g0.z = (unsigned)(gaddr & 0xFFFFFFFFu);                      // global_addr[31:0]
    g0.w = (unsigned)((gaddr >> 32) & 0x01FFFFFFu) | (2u << 30); // addr[56:32] | type=2
    v8i g1;
    g1[0] = (int)((unsigned)data_size_log << 16);                // mask=0, data_size
    g1[1] = (int)((tensor_d0 & 0xFFFFu) << 16);                  // tensor_dim0[15:0]
    g1[2] = (int)((tensor_d0 >> 16) | ((tensor_d1 & 0xFFFFu) << 16));
    g1[3] = (int)((tensor_d1 >> 16) | (tile_d0 << 16));          // tile_dim0
    g1[4] = (int)(tile_d1);                                      // tile_dim1, tile_dim2=0
    g1[5] = (int)(stride0 & 0xFFFFFFFFu);                        // dim0 stride lo
    g1[6] = (int)((stride0 >> 32) & 0xFFFFu);                    // stride hi | dim1 stride=0
    g1[7] = 0;
    v4i z4 = {0, 0, 0, 0};
    v8i z8 = {0, 0, 0, 0, 0, 0, 0, 0};
    __builtin_amdgcn_tensor_load_to_lds(g0, g1, z4, z4, z8, 0);
}

// ---------------- element-wise conversion kernels ----------------
__global__ void f32_to_bf16_kernel(const float* __restrict__ in,
                                   unsigned short* __restrict__ out, int n) {
    int i = blockIdx.x * 256 + threadIdx.x;
    if (i < n) out[i] = f2bf(in[i]);
}

// in: R x C row-major (f32) -> out: C x R row-major (bf16)  (i.e. W^T)
__global__ void transpose_bf16_kernel(const float* __restrict__ in,
                                      unsigned short* __restrict__ out, int R, int C) {
    int i = blockIdx.x * 256 + threadIdx.x;
    if (i < R * C) {
        int r = i / C, c = i % C;
        out[(size_t)c * R + r] = f2bf(in[i]);
    }
}

// ---------------- GEMM: C(f32,MxN) = A(bf16,MxK) @ Bt(bf16,NxK)^T ----------------
// 64x64 block tile per WG; A/B 64x32 panels TDM-staged into LDS, double buffered.
__global__ __launch_bounds__(256)
void gemm_bf16_tdm_kernel(const unsigned short* __restrict__ A,
                          const unsigned short* __restrict__ Bt,
                          float* __restrict__ C, int M, int N, int K) {
    __shared__ unsigned short Abuf[2][64 * 32];
    __shared__ unsigned short Bbuf[2][64 * 32];

    const int lane = threadIdx.x & 31, wave = threadIdx.x >> 5;
    const int nbn = N >> 6;
    const int bm = blockIdx.x / nbn, bn = blockIdx.x % nbn;
    const int NS = K >> 5;

    const unsigned short* Abase = A  + (size_t)bm * 64 * K;
    const unsigned short* Bbase = Bt + (size_t)bn * 64 * K;

    if (wave == 0) {  // prologue: stage k-step 0
        tdm_load_2d(lds_offset(Abuf[0]), (unsigned long long)(size_t)Abase,
                    32, 64, 32, 64, (unsigned long long)K, 1);
        tdm_load_2d(lds_offset(Bbuf[0]), (unsigned long long)(size_t)Bbase,
                    32, 64, 32, 64, (unsigned long long)K, 1);
    }

    const int mw = wave >> 1, nw = wave & 1;
    v8f acc0 = {}, acc1 = {};

    for (int s = 0; s < NS; ++s) {
        const int cur = s & 1;
        if (wave == 0) {
            if (s + 1 < NS) {
                unsigned long long off = (unsigned long long)(s + 1) * 32 * 2; // bytes
                tdm_load_2d(lds_offset(Abuf[cur ^ 1]),
                            (unsigned long long)(size_t)Abase + off,
                            32, 64, 32, 64, (unsigned long long)K, 1);
                tdm_load_2d(lds_offset(Bbuf[cur ^ 1]),
                            (unsigned long long)(size_t)Bbase + off,
                            32, 64, 32, 64, (unsigned long long)K, 1);
                __builtin_amdgcn_s_wait_tensorcnt(2);   // panels for step s done
            } else {
                __builtin_amdgcn_s_wait_tensorcnt(0);
            }
        }
        __syncthreads();
        v16bf a  = load_frag(Abuf[cur] + mw * 16 * 32, 32, lane);
        v16bf b0 = load_frag(Bbuf[cur] + (nw * 32) * 32, 32, lane);
        v16bf b1 = load_frag(Bbuf[cur] + (nw * 32 + 16) * 32, 32, lane);
        acc0 = wmma_bf16(a, b0, acc0);
        acc1 = wmma_bf16(a, b1, acc1);
        __syncthreads();
    }

    const int nl = lane & 15, hi = lane >> 4;
#pragma unroll
    for (int r = 0; r < 8; ++r) {
        int row = bm * 64 + mw * 16 + hi * 8 + r;
        C[(size_t)row * N + bn * 64 + nw * 32 + nl]      = acc0[r];
        C[(size_t)row * N + bn * 64 + nw * 32 + 16 + nl] = acc1[r];
    }
}

// ---------------- chunked causal linear attention (one WG per (b,h)) ----------------
__global__ __launch_bounds__(256)
void attn_scan_kernel(const float* __restrict__ q, const float* __restrict__ k,
                      const float* __restrict__ v, unsigned short* __restrict__ ybf) {
    extern __shared__ char smem[];
    unsigned short* Qf = (unsigned short*)smem;          // 32 x 288
    unsigned short* Kf = Qf + 32 * DPAD;                 // 32 x 288
    unsigned short* Kt = Kf + 32 * DPAD;                 // 288 x 32
    unsigned short* Vb = Kt + DPAD * 32;                 // 64 x 32  (V^T)
    unsigned short* Sb = Vb + HDIM * CHUNK;              // 64 x 288 (S^T bf16)
    unsigned short* Am = Sb + HDIM * DPAD;               // 32 x 32
    float* sumk = (float*)(Am + CHUNK * CHUNK);          // 288
    float* zl   = sumk + DPAD;                           // 32
    float* qs   = zl + 32;                               // 2 x (32x16), TDM dest
    float* ks   = qs + 2 * CHUNK * FD;                   // 2 x (32x16), TDM dest
    float* vraw = ks + 2 * CHUNK * FD;                   // 2 x (32x64), TDM dest

    const int bh = blockIdx.x, b = bh >> 4, h = bh & 15;
    const int tid = threadIdx.x, lane = tid & 31, wave = tid >> 5;

    v8f Sacc[9];
#pragma unroll
    for (int i = 0; i < 9; ++i) Sacc[i] = (v8f){};

    for (int i = tid; i < HDIM * DPAD; i += 256) Sb[i] = 0;
    for (int i = tid; i < DPAD; i += 256) sumk[i] = 0.f;

    const float rrd_inv = 0.5f;
    const float c2 = 0.17677669529663687f;
    const float* qb = q + (size_t)b * LSEQ * (NHEAD * FD) + h * FD;
    const float* kb = k + (size_t)b * LSEQ * (NHEAD * FD) + h * FD;
    const float* vb = v + (size_t)b * LSEQ * (NHEAD * HDIM) + h * HDIM;

    if (wave == 0) {  // prologue: TDM-stage chunk 0 into buffer 0
        tdm_load_2d(lds_offset(qs), (unsigned long long)(size_t)qb,
                    FD, CHUNK, FD, CHUNK, NHEAD * FD, 2);
        tdm_load_2d(lds_offset(ks), (unsigned long long)(size_t)kb,
                    FD, CHUNK, FD, CHUNK, NHEAD * FD, 2);
        tdm_load_2d(lds_offset(vraw), (unsigned long long)(size_t)vb,
                    HDIM, CHUNK, HDIM, CHUNK, NHEAD * HDIM, 2);
    }
    __syncthreads();

    for (int c = 0; c < NCHUNK; ++c) {
        const int l0 = c * CHUNK;
        const int cb = c & 1;
        const float* qsb = qs + cb * CHUNK * FD;
        const float* ksb = ks + cb * CHUNK * FD;
        const float* vrb = vraw + cb * CHUNK * HDIM;

        if (wave == 0) __builtin_amdgcn_s_wait_tensorcnt(0);  // chunk c staged
        __syncthreads();

        // phase 1: Taylor features + V^T bf16 conversion
        {
            int r = tid >> 3, sub = tid & 7;
#pragma unroll 4
            for (int dd = 0; dd < 36; ++dd) {
                int d = sub * 36 + dd;
                float qv, kv;
                if (d == 0)       { qv = 1.f; kv = 1.f; }
                else if (d < 17)  { qv = qsb[r * FD + d - 1] * rrd_inv;
                                    kv = ksb[r * FD + d - 1] * rrd_inv; }
                else if (d < 273) { int e = d - 17, ii = e >> 4, jj = e & 15;
                                    qv = qsb[r * FD + ii] * qsb[r * FD + jj] * c2;
                                    kv = ksb[r * FD + ii] * ksb[r * FD + jj] * c2; }
                else              { qv = 0.f; kv = 0.f; }
                unsigned short qh = f2bf(qv), kh = f2bf(kv);
                Qf[r * DPAD + d] = qh;
                Kf[r * DPAD + d] = kh;
                Kt[d * CHUNK + r] = kh;
            }
            for (int i = tid; i < CHUNK * HDIM; i += 256) {
                int rr = i / HDIM, col = i % HDIM;
                Vb[col * CHUNK + rr] = f2bf(vrb[rr * HDIM + col]);
            }
        }
        __syncthreads();

        // overlap: TDM-stage chunk c+1 while phases 2..6 run on chunk c
        if (wave == 0 && c + 1 < NCHUNK) {
            int nb = (c + 1) & 1;
            const float* qn = qb + (size_t)(l0 + CHUNK) * (NHEAD * FD);
            const float* kn = kb + (size_t)(l0 + CHUNK) * (NHEAD * FD);
            const float* vn = vb + (size_t)(l0 + CHUNK) * (NHEAD * HDIM);
            tdm_load_2d(lds_offset(qs + nb * CHUNK * FD),
                        (unsigned long long)(size_t)qn, FD, CHUNK, FD, CHUNK, NHEAD * FD, 2);
            tdm_load_2d(lds_offset(ks + nb * CHUNK * FD),
                        (unsigned long long)(size_t)kn, FD, CHUNK, FD, CHUNK, NHEAD * FD, 2);
            tdm_load_2d(lds_offset(vraw + nb * CHUNK * HDIM),
                        (unsigned long long)(size_t)vn, HDIM, CHUNK, HDIM, CHUNK, NHEAD * HDIM, 2);
        }

        // phase 2: A = tril(Qf @ Kf^T)  (waves 0..3)
        if (wave < 4) {
            int mt = wave >> 1, nt = wave & 1;
            v8f acc = {};
            if (mt >= nt) {
                for (int kk = 0; kk < DPAD / 32; ++kk) {
                    v16bf a  = load_frag(Qf + mt * 16 * DPAD + kk * 32, DPAD, lane);
                    v16bf bb = load_frag(Kf + nt * 16 * DPAD + kk * 32, DPAD, lane);
                    acc = wmma_bf16(a, bb, acc);
                }
            }
            int nl = lane & 15, hi = lane >> 4;
#pragma unroll
            for (int r = 0; r < 8; ++r) {
                int gm = mt * 16 + hi * 8 + r, gn = nt * 16 + nl;
                Am[gm * CHUNK + gn] = f2bf((gm >= gn) ? acc[r] : 0.f);
            }
        }
        __syncthreads();

        // phase 3: z_t = 1/(qf_t.sumk_prev + rowsum(tril A) + eps)
        if (wave == 0) {
            int r = lane;
            float s = 1e-12f;
            for (int d = 0; d < 273; ++d) s += bf2f(Qf[r * DPAD + d]) * sumk[d];
            for (int j = 0; j < CHUNK; ++j) s += bf2f(Am[r * CHUNK + j]);
            zl[r] = 1.f / s;
        }
        __syncthreads();

        // phase 4: sumk += column sums of Kf
        for (int d = tid; d < DPAD; d += 256) {
            float a = sumk[d];
            for (int r = 0; r < CHUNK; ++r) a += bf2f(Kf[r * DPAD + d]);
            sumk[d] = a;
        }

        // phase 5: y = (Qf @ S_prev + A @ V) * z   (8 waves, 2x4 tiles of 32x64)
        {
            int mt = wave >> 2, nt = wave & 3;
            v8f acc = {};
            for (int kk = 0; kk < DPAD / 32; ++kk) {
                v16bf a  = load_frag(Qf + mt * 16 * DPAD + kk * 32, DPAD, lane);
                v16bf bb = load_frag(Sb + nt * 16 * DPAD + kk * 32, DPAD, lane);
                acc = wmma_bf16(a, bb, acc);
            }
            {
                v16bf a  = load_frag(Am + mt * 16 * CHUNK, CHUNK, lane);
                v16bf bb = load_frag(Vb + nt * 16 * CHUNK, CHUNK, lane);
                acc = wmma_bf16(a, bb, acc);
            }
            int nl = lane & 15, hi = lane >> 4;
#pragma unroll
            for (int r = 0; r < 8; ++r) {
                int ml = hi * 8 + r;
                int row = l0 + mt * 16 + ml;
                float val = acc[r] * zl[mt * 16 + ml];
                ybf[(size_t)(b * LSEQ + row) * (NHEAD * HDIM) + h * HDIM + nt * 16 + nl] =
                    f2bf(val);
            }
        }
        __syncthreads();   // all reads of Sb done before refresh

        // phase 6: S += Kf^T @ V (VGPR accumulators) ; refresh bf16 shadow Sb
        {
            int nl = lane & 15, hi = lane >> 4;
#pragma unroll
            for (int i = 0; i < 9; ++i) {
                int t = wave + 8 * i;
                int mt = t >> 2, nt = t & 3;
                v16bf a  = load_frag(Kt + mt * 16 * CHUNK, CHUNK, lane);
                v16bf bb = load_frag(Vb + nt * 16 * CHUNK, CHUNK, lane);
                Sacc[i] = wmma_bf16(a, bb, Sacc[i]);
                unsigned short* dst = Sb + (size_t)(nt * 16 + nl) * DPAD + mt * 16 + hi * 8;
#pragma unroll
                for (int r = 0; r < 8; ++r) dst[r] = f2bf(Sacc[i][r]);
            }
        }
        __syncthreads();
    }
}

// ---------------- host ----------------
extern "C" void kernel_launch(void* const* d_in, const int* in_sizes, int n_in,
                              void* d_out, int out_size, void* d_ws, size_t ws_size,
                              hipStream_t stream) {
    const float* x  = (const float*)d_in[0];   // (2,2048,1024)
    const float* Wq = (const float*)d_in[1];   // (1024,256)
    const float* Wk = (const float*)d_in[2];   // (1024,256)
    const float* Wv = (const float*)d_in[3];   // (1024,1024)
    const float* Wo = (const float*)d_in[4];   // (1024,1024)
    float* out = (float*)d_out;                // (2,2048,1024)

    const int M = 2 * LSEQ;        // 4096 rows
    const int DM = 1024;
    const int NQK = NHEAD * FD;    // 256

    char* w = (char*)d_ws;
    unsigned short* xb  = (unsigned short*)w; w += (size_t)M * DM * 2;
    unsigned short* WqT = (unsigned short*)w; w += (size_t)NQK * DM * 2;
    unsigned short* WkT = (unsigned short*)w; w += (size_t)NQK * DM * 2;
    unsigned short* WvT = (unsigned short*)w; w += (size_t)DM * DM * 2;
    unsigned short* WoT = (unsigned short*)w; w += (size_t)DM * DM * 2;
    float* qbuf = (float*)w; w += (size_t)M * NQK * 4;
    float* kbuf = (float*)w; w += (size_t)M * NQK * 4;
    float* vbuf = (float*)w; w += (size_t)M * DM * 4;
    unsigned short* ybf = (unsigned short*)w; w += (size_t)M * DM * 2;

    // conversions
    f32_to_bf16_kernel<<<(M * DM + 255) / 256, 256, 0, stream>>>(x, xb, M * DM);
    transpose_bf16_kernel<<<(DM * NQK + 255) / 256, 256, 0, stream>>>(Wq, WqT, DM, NQK);
    transpose_bf16_kernel<<<(DM * NQK + 255) / 256, 256, 0, stream>>>(Wk, WkT, DM, NQK);
    transpose_bf16_kernel<<<(DM * DM + 255) / 256, 256, 0, stream>>>(Wv, WvT, DM, DM);
    transpose_bf16_kernel<<<(DM * DM + 255) / 256, 256, 0, stream>>>(Wo, WoT, DM, DM);

    // projections: q = x@Wq, k = x@Wk, v = x@Wv  (TDM-staged block GEMM)
    gemm_bf16_tdm_kernel<<<(M / 64) * (NQK / 64), 256, 0, stream>>>(xb, WqT, qbuf, M, NQK, DM);
    gemm_bf16_tdm_kernel<<<(M / 64) * (NQK / 64), 256, 0, stream>>>(xb, WkT, kbuf, M, NQK, DM);
    gemm_bf16_tdm_kernel<<<(M / 64) * (DM / 64), 256, 0, stream>>>(xb, WvT, vbuf, M, DM, DM);

    // chunked causal linear-attention scan, one workgroup per (b,h)
    const size_t smem_bytes =
        (size_t)(32 * DPAD + 32 * DPAD + DPAD * 32 + HDIM * CHUNK + HDIM * DPAD +
                 CHUNK * CHUNK) * 2 +
        (size_t)(DPAD + 32 + 2 * CHUNK * FD + 2 * CHUNK * FD + 2 * CHUNK * HDIM) * 4;
    (void)hipFuncSetAttribute((const void*)attn_scan_kernel,
                              hipFuncAttributeMaxDynamicSharedMemorySize, (int)smem_bytes);
    attn_scan_kernel<<<2 * NHEAD, 256, smem_bytes, stream>>>(qbuf, kbuf, vbuf, ybf);

    // output projection: out = y @ Wo
    gemm_bf16_tdm_kernel<<<(M / 64) * (DM / 64), 256, 0, stream>>>(ybf, WoT, out, M, DM, DM);
}
